// GlobalContextAttentionBlock_60739427500843
// MI455X (gfx1250) — matrix operand order repaired and checked by reading, not compiled, not run on hardware
//
#include <hip/hip_runtime.h>
#include <hip/hip_bf16.h>
#include <stdint.h>

// ---------------- problem constants ----------------
#define NPIX 16384      // H*W
#define CINC 256
#define CC   128
#define NBAT 8
#define LDK  136        // LDS row stride (ushorts): 272B rows -> 16B aligned, low bank conflict

// workspace layout (bytes)
#define OFF_WBF   0u                 // 131072 ushort = 262144 B (bf16 weights)
#define OFF_BNP   262144u            // 5 layers * 256 f32 (scale|shift)       = 5120 B
#define OFF_Z     268288u            // 8*128 f32                              = 4096 B
#define OFF_SF    272384u            // 8*128*128 f32                          = 524288 B
#define OFF_SBF   797696u            // 8*128*128 bf16                         = 262144 B
#define OFF_KEY   1060864u           // 8*128*16384 bf16                       = 33554432 B
#define OFF_VAL   34615296u          // 8*128*16384 bf16                       = 33554432 B
// total ~68.2 MB

// bf16 weight sub-offsets (ushort units): layers q1,q2,k1,k2,v
#define WO_Q1 0
#define WO_Q2 32768
#define WO_K1 49152
#define WO_K2 81920
#define WO_V  98304

typedef __attribute__((ext_vector_type(16))) __bf16 v16bf;
typedef __attribute__((ext_vector_type(8)))  float  v8f;

union FragAB { v16bf v; uint4 q[2]; };

__device__ __forceinline__ unsigned short f2bf(float x) {
  unsigned u = __float_as_uint(x);
  unsigned r = u + 0x7FFFu + ((u >> 16) & 1u);   // round-to-nearest-even
  return (unsigned short)(r >> 16);
}
__device__ __forceinline__ float bf2f(unsigned short h) {
  return __uint_as_float(((unsigned)h) << 16);
}
__device__ __forceinline__ float softplusf(float x) {
  return fmaxf(x, 0.0f) + __logf(1.0f + __expf(-fabsf(x)));
}

// A fragment (16x32 bf16) from row-major [M][ldk] storage.
// lanes 0-15: K = k0..k0+15; lanes 16-31: K = k0+16..k0+31 (16 contiguous bf16).
__device__ __forceinline__ v16bf load_a_rowmajor(const unsigned short* w, int ldk,
                                                 int m, int k0, int hi) {
  const unsigned short* p = w + (size_t)m * (size_t)ldk + (size_t)(k0 + hi * 16);
  FragAB f;
  f.q[0] = *(const uint4*)(p);
  f.q[1] = *(const uint4*)(p + 8);
  return f.v;
}
// B fragment (32x16 bf16) from K-contiguous storage Bt[n][k] (row stride ldk).
// lanes 0-15: K = k0..k0+15; lanes 16-31: K = k0+16..k0+31 (16 contiguous bf16).
__device__ __forceinline__ v16bf load_b_kcontig(const unsigned short* bt, int ldk,
                                                int n, int k0, int hi) {
  const unsigned short* p = bt + (size_t)n * (size_t)ldk + (size_t)(k0 + hi * 16);
  FragAB f;
  f.q[0] = *(const uint4*)(p);
  f.q[1] = *(const uint4*)(p + 8);
  return f.v;
}
__device__ __forceinline__ v8f wmma_bf16(v16bf a, v16bf b, v8f c) {
  return __builtin_amdgcn_wmma_f32_16x16x32_bf16(false, a, false, b, (short)0, c,
                                                 false, false);
}

// Stage a 128(K) x 128(N) f32 global tile into LDS transposed as bf16 [n][k], stride LDK.
__device__ __forceinline__ void load_tile_tr(const float* x, int c0, int n0,
                                             unsigned short* buf, int tid) {
  const int lane = tid & 31, w = tid >> 5;
  for (int r = w; r < 128; r += 8) {
    const float* src = x + (size_t)(c0 + r) * NPIX + (size_t)(n0 + lane * 4);
    float4 v = *(const float4*)src;
    unsigned short* d = buf + (size_t)(lane * 4) * LDK + r;
    d[0 * LDK] = f2bf(v.x);
    d[1 * LDK] = f2bf(v.y);
    d[2 * LDK] = f2bf(v.z);
    d[3 * LDK] = f2bf(v.w);
  }
}

// BN + activation on D fragments, store transposed into LDS buf[n][m] (bf16), b128 stores.
__device__ __forceinline__ void bn_act_store_tr(const v8f* acc, const float* sc,
                                                const float* sh, int act,
                                                unsigned short* buf, int m0,
                                                int lane, int hi) {
  const int nl = lane & 15;
#pragma unroll
  for (int nt = 0; nt < 8; ++nt) {
    unsigned short t[8];
#pragma unroll
    for (int j = 0; j < 8; ++j) {
      float y = acc[nt][j] * sc[j] + sh[j];
      y = act ? softplusf(y) : fmaxf(y, 0.0f);
      t[j] = f2bf(y);
    }
    uint4 pk;
    pk.x = (unsigned)t[0] | ((unsigned)t[1] << 16);
    pk.y = (unsigned)t[2] | ((unsigned)t[3] << 16);
    pk.z = (unsigned)t[4] | ((unsigned)t[5] << 16);
    pk.w = (unsigned)t[6] | ((unsigned)t[7] << 16);
    *(uint4*)(buf + (size_t)(nt * 16 + nl) * LDK + (m0 + hi * 8)) = pk;
  }
}

// BN + activation on D fragments, store row-major bf16 [m][NPIX] to global (batch base applied).
__device__ __forceinline__ void bn_act_store_g(const v8f* acc, const float* sc,
                                               const float* sh, int act,
                                               unsigned short* out, int m0, int n0,
                                               int lane, int hi) {
  const int nl = lane & 15;
#pragma unroll
  for (int nt = 0; nt < 8; ++nt) {
#pragma unroll
    for (int j = 0; j < 8; ++j) {
      float y = acc[nt][j] * sc[j] + sh[j];
      y = act ? softplusf(y) : fmaxf(y, 0.0f);
      out[(size_t)(m0 + j + 8 * hi) * NPIX + (size_t)(n0 + nt * 16 + nl)] = f2bf(y);
    }
  }
}

__device__ __forceinline__ void load_bn8(const float* bnp, int layer, int m0, int hi,
                                         float* sc, float* sh) {
#pragma unroll
  for (int j = 0; j < 8; ++j) {
    int m = m0 + j + 8 * hi;
    sc[j] = bnp[layer * 256 + m];
    sh[j] = bnp[layer * 256 + 128 + m];
  }
}

// ---------------- prep: weights -> bf16, BN -> scale/shift ----------------
struct PrepArgs {
  const float* w[5];     // q1,q2,k1,k2,v
  const float* bn[20];   // {g,b,m,v} x {q1,q2,k1,k2,v}
  unsigned short* wbf;
  float* bnp;
};

__global__ __launch_bounds__(256) void prep_kernel(PrepArgs a) {
  int i = blockIdx.x * 256 + threadIdx.x;
  if (i < 131072) {
    const float* w; int rel;
    if      (i < 32768)  { w = a.w[0]; rel = i; }
    else if (i < 49152)  { w = a.w[1]; rel = i - 32768; }
    else if (i < 81920)  { w = a.w[2]; rel = i - 49152; }
    else if (i < 98304)  { w = a.w[3]; rel = i - 81920; }
    else                 { w = a.w[4]; rel = i - 98304; }
    a.wbf[i] = f2bf(w[rel]);
  } else if (i < 131072 + 640) {
    int idx = i - 131072, layer = idx >> 7, c = idx & 127;
    float g = a.bn[layer * 4 + 0][c], b = a.bn[layer * 4 + 1][c];
    float m = a.bn[layer * 4 + 2][c], v = a.bn[layer * 4 + 3][c];
    float sc = g * rsqrtf(v + 1e-5f);
    a.bnp[layer * 256 + c] = sc;
    a.bnp[layer * 256 + 128 + c] = b - m * sc;
  }
}

__global__ __launch_bounds__(256) void zero_s(float* S) {
  int i = blockIdx.x * 256 + threadIdx.x;
  if (i < NBAT * CC * CC) S[i] = 0.0f;
}

// ---------------- key/value producer ----------------
__global__ __launch_bounds__(256) void kv_kernel(const float* __restrict__ src,
                                                 const unsigned short* __restrict__ wbf,
                                                 const float* __restrict__ bnp,
                                                 unsigned short* __restrict__ keyo,
                                                 unsigned short* __restrict__ valo) {
  __shared__ unsigned short buf[128 * LDK];
  const int b = blockIdx.y, n0 = blockIdx.x * 128;
  const int tid = threadIdx.x, lane = tid & 31, w = tid >> 5;
  const int hi = lane >> 4, lm = lane & 15, m0 = w * 16;
  const float* xb = src + (size_t)b * CINC * NPIX;

  v8f acc1[8] = {};  // k1 path
  v8f accv[8] = {};  // v path
  for (int ph = 0; ph < 2; ++ph) {
    load_tile_tr(xb, ph * 128, n0, buf, tid);
    __syncthreads();
#pragma unroll
    for (int ks = 0; ks < 4; ++ks) {
      v16bf a1 = load_a_rowmajor(wbf + WO_K1, CINC, m0 + lm, ph * 128 + ks * 32, hi);
      v16bf a2 = load_a_rowmajor(wbf + WO_V,  CINC, m0 + lm, ph * 128 + ks * 32, hi);
#pragma unroll
      for (int nt = 0; nt < 8; ++nt) {
        v16bf bf = load_b_kcontig(buf, LDK, nt * 16 + lm, ks * 32, hi);
        acc1[nt] = wmma_bf16(a1, bf, acc1[nt]);
        accv[nt] = wmma_bf16(a2, bf, accv[nt]);
      }
    }
    __syncthreads();
  }

  float sc[8], sh[8];
  load_bn8(bnp, 4, m0, hi, sc, sh);                       // v: bn + relu -> value
  bn_act_store_g(accv, sc, sh, 0, valo + (size_t)b * CC * NPIX, m0, n0, lane, hi);

  load_bn8(bnp, 2, m0, hi, sc, sh);                       // k1: bn + relu -> LDS (transposed)
  bn_act_store_tr(acc1, sc, sh, 0, buf, m0, lane, hi);
  __syncthreads();

  v8f acc2[8] = {};                                       // k2 GEMM (K=128)
#pragma unroll
  for (int ks = 0; ks < 4; ++ks) {
    v16bf a = load_a_rowmajor(wbf + WO_K2, CC, m0 + lm, ks * 32, hi);
#pragma unroll
    for (int nt = 0; nt < 8; ++nt) {
      v16bf bf = load_b_kcontig(buf, LDK, nt * 16 + lm, ks * 32, hi);
      acc2[nt] = wmma_bf16(a, bf, acc2[nt]);
    }
  }
  load_bn8(bnp, 3, m0, hi, sc, sh);                       // k2: bn + softplus -> key
  bn_act_store_g(acc2, sc, sh, 1, keyo + (size_t)b * CC * NPIX, m0, n0, lane, hi);
}

// ---------------- S = value . key^T (atomic K-chunked) ----------------
__global__ __launch_bounds__(256) void s_kernel(const unsigned short* __restrict__ val,
                                                const unsigned short* __restrict__ key,
                                                float* __restrict__ S) {
  const int b = blockIdx.y, k0g = blockIdx.x * 512;
  const int tid = threadIdx.x, lane = tid & 31, w = tid >> 5;
  const int hi = lane >> 4, lm = lane & 15, m0 = w * 16;
  const unsigned short* vb = val + (size_t)b * CC * NPIX;
  const unsigned short* kb = key + (size_t)b * CC * NPIX;

  v8f acc[8] = {};
  for (int ks = 0; ks < 16; ++ks) {
    v16bf a = load_a_rowmajor(vb, NPIX, m0 + lm, k0g + ks * 32, hi);
#pragma unroll
    for (int nt = 0; nt < 8; ++nt) {
      v16bf bf = load_b_kcontig(kb, NPIX, nt * 16 + lm, k0g + ks * 32, hi);
      acc[nt] = wmma_bf16(a, bf, acc[nt]);
    }
  }
  float* Sb = S + (size_t)b * CC * CC;
#pragma unroll
  for (int nt = 0; nt < 8; ++nt)
#pragma unroll
    for (int j = 0; j < 8; ++j)
      atomicAdd(Sb + (size_t)(m0 + j + 8 * hi) * CC + (nt * 16 + lm), acc[nt][j]);
}

// ---------------- Z[b][d] = sum_n key ----------------
__global__ __launch_bounds__(256) void z_kernel(const unsigned short* __restrict__ key,
                                                float* __restrict__ Z) {
  __shared__ float red[256];
  const int row = blockIdx.x;  // b*128 + d
  const uint4* kr = (const uint4*)(key + (size_t)row * NPIX);
  float s = 0.0f;
  for (int i = threadIdx.x; i < NPIX / 8; i += 256) {
    uint4 q = kr[i];
    s += bf2f((unsigned short)(q.x & 0xFFFF)) + bf2f((unsigned short)(q.x >> 16));
    s += bf2f((unsigned short)(q.y & 0xFFFF)) + bf2f((unsigned short)(q.y >> 16));
    s += bf2f((unsigned short)(q.z & 0xFFFF)) + bf2f((unsigned short)(q.z >> 16));
    s += bf2f((unsigned short)(q.w & 0xFFFF)) + bf2f((unsigned short)(q.w >> 16));
  }
  red[threadIdx.x] = s;
  __syncthreads();
  for (int off = 128; off > 0; off >>= 1) {
    if (threadIdx.x < off) red[threadIdx.x] += red[threadIdx.x + off];
    __syncthreads();
  }
  if (threadIdx.x == 0) Z[row] = red[0];
}

__global__ __launch_bounds__(256) void sconv_kernel(const float* __restrict__ S,
                                                    unsigned short* __restrict__ Sbf) {
  int i = blockIdx.x * 256 + threadIdx.x;
  if (i < NBAT * CC * CC) Sbf[i] = f2bf(S[i]);
}

// ---------------- query + final attention output ----------------
__global__ __launch_bounds__(256) void out_kernel(const float* __restrict__ tgt,
                                                  const unsigned short* __restrict__ wbf,
                                                  const float* __restrict__ bnp,
                                                  const unsigned short* __restrict__ Sbf,
                                                  const float* __restrict__ Z,
                                                  float* __restrict__ out) {
  __shared__ unsigned short buf[128 * LDK];
  __shared__ float zsh[128];
  __shared__ float denom[128];
  const int b = blockIdx.y, n0 = blockIdx.x * 128;
  const int tid = threadIdx.x, lane = tid & 31, w = tid >> 5;
  const int hi = lane >> 4, lm = lane & 15, m0 = w * 16;
  const float* xb = tgt + (size_t)b * CINC * NPIX;
  if (tid < 128) zsh[tid] = Z[b * 128 + tid];

  v8f acc1[8] = {};                                       // q1 GEMM (K=256, two phases)
  for (int ph = 0; ph < 2; ++ph) {
    load_tile_tr(xb, ph * 128, n0, buf, tid);
    __syncthreads();
#pragma unroll
    for (int ks = 0; ks < 4; ++ks) {
      v16bf a = load_a_rowmajor(wbf + WO_Q1, CINC, m0 + lm, ph * 128 + ks * 32, hi);
#pragma unroll
      for (int nt = 0; nt < 8; ++nt) {
        v16bf bf = load_b_kcontig(buf, LDK, nt * 16 + lm, ks * 32, hi);
        acc1[nt] = wmma_bf16(a, bf, acc1[nt]);
      }
    }
    __syncthreads();
  }

  float sc[8], sh[8];
  load_bn8(bnp, 0, m0, hi, sc, sh);                       // q1: bn + relu -> LDS
  bn_act_store_tr(acc1, sc, sh, 0, buf, m0, lane, hi);
  __syncthreads();

  v8f acc2[8] = {};                                       // q2 GEMM (K=128)
#pragma unroll
  for (int ks = 0; ks < 4; ++ks) {
    v16bf a = load_a_rowmajor(wbf + WO_Q2, CC, m0 + lm, ks * 32, hi);
#pragma unroll
    for (int nt = 0; nt < 8; ++nt) {
      v16bf bf = load_b_kcontig(buf, LDK, nt * 16 + lm, ks * 32, hi);
      acc2[nt] = wmma_bf16(a, bf, acc2[nt]);
    }
  }
  __syncthreads();                                        // all q2 reads of buf done
  load_bn8(bnp, 1, m0, hi, sc, sh);                       // q2: bn + softplus -> query in LDS [n][d]
  bn_act_store_tr(acc2, sc, sh, 1, buf, m0, lane, hi);
  __syncthreads();

  if (tid < 128) {                                        // denom[n] = max(sum_d Z[d]*q[d][n], eps)
    float s = 0.0f;
    const unsigned short* qr = buf + (size_t)tid * LDK;
    for (int d = 0; d < 128; ++d) s += zsh[d] * bf2f(qr[d]);
    denom[tid] = fmaxf(s, 1e-12f);
  }
  __syncthreads();

  v8f acc3[8] = {};                                       // out = S . query (K=128)
#pragma unroll
  for (int ks = 0; ks < 4; ++ks) {
    v16bf a = load_a_rowmajor(Sbf + (size_t)b * CC * CC, CC, m0 + lm, ks * 32, hi);
#pragma unroll
    for (int nt = 0; nt < 8; ++nt) {
      v16bf bf = load_b_kcontig(buf, LDK, nt * 16 + lm, ks * 32, hi);
      acc3[nt] = wmma_bf16(a, bf, acc3[nt]);
    }
  }
  float* ob = out + (size_t)b * CC * NPIX;
#pragma unroll
  for (int nt = 0; nt < 8; ++nt) {
    float inv = 1.0f / denom[nt * 16 + lm];
#pragma unroll
    for (int j = 0; j < 8; ++j)
      ob[(size_t)(m0 + j + 8 * hi) * NPIX + (size_t)(n0 + nt * 16 + lm)] =
          acc3[nt][j] * inv;
  }
}

extern "C" void kernel_launch(void* const* d_in, const int* in_sizes, int n_in,
                              void* d_out, int out_size, void* d_ws, size_t ws_size,
                              hipStream_t stream) {
  (void)in_sizes; (void)n_in; (void)out_size; (void)ws_size;
  const float* tgt = (const float*)d_in[0];
  const float* src = (const float*)d_in[1];

  char* ws = (char*)d_ws;
  unsigned short* wbf  = (unsigned short*)(ws + OFF_WBF);
  float*          bnp  = (float*)(ws + OFF_BNP);
  float*          Zb   = (float*)(ws + OFF_Z);
  float*          Sf   = (float*)(ws + OFF_SF);
  unsigned short* Sbf  = (unsigned short*)(ws + OFF_SBF);
  unsigned short* keyb = (unsigned short*)(ws + OFF_KEY);
  unsigned short* valb = (unsigned short*)(ws + OFF_VAL);

  PrepArgs pa;
  for (int i = 0; i < 5; ++i)  pa.w[i]  = (const float*)d_in[2 + i];
  for (int i = 0; i < 20; ++i) pa.bn[i] = (const float*)d_in[7 + i];
  pa.wbf = wbf; pa.bnp = bnp;

  prep_kernel<<<516, 256, 0, stream>>>(pa);
  zero_s<<<512, 256, 0, stream>>>(Sf);
  kv_kernel<<<dim3(128, 8), 256, 0, stream>>>(src, wbf, bnp, keyb, valb);
  s_kernel<<<dim3(32, 8), 256, 0, stream>>>(valb, keyb, Sf);
  z_kernel<<<1024, 256, 0, stream>>>(keyb, Zb);
  sconv_kernel<<<512, 256, 0, stream>>>(Sf, Sbf);
  out_kernel<<<dim3(128, 8), 256, 0, stream>>>(tgt, wbf, bnp, Sbf, Zb, (float*)d_out);
}
